// LightGlue_47476568490216
// MI455X (gfx1250) — compile-verified, matrix operand order checked by reading, not run on hardware
//
#include <hip/hip_runtime.h>
#include <hip/hip_bf16.h>

// ---------------------------------------------------------------------------
// LightGlue forward for MI455X (gfx1250).
// All GEMM-shaped work -> v_wmma_f32_16x16x32_f16 (wave32), operands held in
// f16 end-to-end (weights pre-converted, activation producers emit f16), so
// the GEMM inner loop is pure global_load_b128 + 4x WMMA (A fragment reused
// across a 16x64 output tile). f32 only where VALU math needs it.
// ---------------------------------------------------------------------------

typedef __attribute__((ext_vector_type(16))) _Float16 v16h;
typedef __attribute__((ext_vector_type(8)))  _Float16 v8h;
typedef __attribute__((ext_vector_type(8)))  float    v8f;

#define KB   2        // batch
#define MT   1024     // tokens per image
#define DD   256      // model dim
#define NH   4        // heads
#define HDIM 64       // head dim
#define NL   9        // layers

__device__ __forceinline__ float lsig(float x) {
    return fminf(x, 0.0f) - log1pf(expf(-fabsf(x)));   // stable log-sigmoid
}

// ---------------------------------------------------------------------------
// Batched WMMA GEMM:  C[b] = resid[b] + scale * (A[b] (MxK) @ B[b]^T + bias)
// A, B in f16 ([MxK] and [NxK] row-major). One wave -> 16(M) x 64(N) tile.
// C (f32) and Ch (f16 mirror) are each optional.
// Fragment layout per CDNA5 ISA 7.12.2: lane half selects K sub-groups
// {0..7,16..23} vs {8..15,24..31}; each 8-half run is one 16B load.
// ---------------------------------------------------------------------------
__global__ __launch_bounds__(32)
void wmma_gemm_kernel(const _Float16* __restrict__ A, const _Float16* __restrict__ Bm,
                      const float* __restrict__ bias, const float* __restrict__ resid,
                      float* __restrict__ C, _Float16* __restrict__ Ch,
                      int Msz, int Nsz, int Ksz,
                      long long sAb, long long sBb, long long sCb,
                      float scale)
{
    const int lane = threadIdx.x;
    const int tM = blockIdx.x * 16;
    const int tN = blockIdx.y * 64;
    const long long bz = blockIdx.z;
    A  += bz * sAb;
    Bm += bz * sBb;
    if (C)     C     += bz * sCb;
    if (Ch)    Ch    += bz * sCb;
    if (resid) resid += bz * sCb;

    const int r  = lane & 15;   // A row / W row within N sub-tile
    const int hi = lane >> 4;   // lane half -> K sub-group select

    const _Float16* arow  = A  + (long long)(tM + r) * Ksz;
    const _Float16* brow0 = Bm + (long long)(tN + r) * Ksz;

    v8f acc0 = {}, acc1 = {}, acc2 = {}, acc3 = {};

    for (int k0 = 0; k0 < Ksz; k0 += 32) {
        const v8h a0 = *(const v8h*)(arow + k0 + 8 * hi);
        const v8h a1 = *(const v8h*)(arow + k0 + 16 + 8 * hi);
        const v16h af = __builtin_shufflevector(a0, a1,
            0, 1, 2, 3, 4, 5, 6, 7, 8, 9, 10, 11, 12, 13, 14, 15);
#pragma unroll
        for (int nt = 0; nt < 4; ++nt) {
            const _Float16* brow = brow0 + (long long)(nt * 16) * Ksz;
            const v8h b0 = *(const v8h*)(brow + k0 + 8 * hi);
            const v8h b1 = *(const v8h*)(brow + k0 + 16 + 8 * hi);
            const v16h bf = __builtin_shufflevector(b0, b1,
                0, 1, 2, 3, 4, 5, 6, 7, 8, 9, 10, 11, 12, 13, 14, 15);
            if (nt == 0) acc0 = __builtin_amdgcn_wmma_f32_16x16x32_f16(false, af, false, bf, (short)0, acc0, false, false);
            if (nt == 1) acc1 = __builtin_amdgcn_wmma_f32_16x16x32_f16(false, af, false, bf, (short)0, acc1, false, false);
            if (nt == 2) acc2 = __builtin_amdgcn_wmma_f32_16x16x32_f16(false, af, false, bf, (short)0, acc2, false, false);
            if (nt == 3) acc3 = __builtin_amdgcn_wmma_f32_16x16x32_f16(false, af, false, bf, (short)0, acc3, false, false);
        }
    }

#pragma unroll
    for (int nt = 0; nt < 4; ++nt) {
        const v8f& acc = (nt == 0) ? acc0 : (nt == 1) ? acc1 : (nt == 2) ? acc2 : acc3;
        const int n = tN + nt * 16 + (lane & 15);
        const float bv = bias ? bias[n] : 0.0f;
#pragma unroll
        for (int rr = 0; rr < 8; ++rr) {
            const int m = tM + rr + 8 * hi;
            float v = (acc[rr] + bv) * scale;
            if (resid) v += resid[(long long)m * Nsz + n];
            if (C)  C [(long long)m * Nsz + n] = v;
            if (Ch) Ch[(long long)m * Nsz + n] = (_Float16)v;
        }
    }
}

// generic f32 -> f16 conversion (for weights / initial descriptors)
__global__ void cvt_f16_kernel(const float* __restrict__ X, _Float16* __restrict__ Y,
                               size_t n)
{
    const size_t i = (size_t)blockIdx.x * blockDim.x + threadIdx.x;
    if (i < n) Y[i] = (_Float16)X[i];
}

// ---------------------------------------------------------------------------
// Positional encoding: normalize kpts (per batch min/max) then cos/sin(k @ Wr^T)
// enc layout: [2 (cos,sin)][B][MT][64]
// ---------------------------------------------------------------------------
__global__ void posenc_kernel(const float* __restrict__ kpts,
                              const float* __restrict__ Wr,
                              float* __restrict__ enc)
{
    const int b = blockIdx.x, t = threadIdx.x;
    const float* kp = kpts + (size_t)b * MT * 2;
    __shared__ float rmn[2][256], rmx[2][256];
    float mnx = 3.4e38f, mny = 3.4e38f, mxx = -3.4e38f, mxy = -3.4e38f;
    for (int n = t; n < MT; n += 256) {
        float x = kp[n * 2], y = kp[n * 2 + 1];
        mnx = fminf(mnx, x); mxx = fmaxf(mxx, x);
        mny = fminf(mny, y); mxy = fmaxf(mxy, y);
    }
    rmn[0][t] = mnx; rmn[1][t] = mny; rmx[0][t] = mxx; rmx[1][t] = mxy;
    __syncthreads();
    for (int k = 128; k > 0; k >>= 1) {
        if (t < k) {
            rmn[0][t] = fminf(rmn[0][t], rmn[0][t + k]);
            rmn[1][t] = fminf(rmn[1][t], rmn[1][t + k]);
            rmx[0][t] = fmaxf(rmx[0][t], rmx[0][t + k]);
            rmx[1][t] = fmaxf(rmx[1][t], rmx[1][t + k]);
        }
        __syncthreads();
    }
    const float szx = 1.0f + rmx[0][0] - rmn[0][0];
    const float szy = 1.0f + rmx[1][0] - rmn[1][0];
    const float shx = szx * 0.5f, shy = szy * 0.5f;
    const float scl = fmaxf(szx, szy) * 0.5f;
    for (int n = t; n < MT; n += 256) {
        const float x = (kp[n * 2]     - shx) / scl;
        const float y = (kp[n * 2 + 1] - shy) / scl;
        float* ec = enc + ((size_t)b * MT + n) * 64;                   // cos plane
        float* es = enc + ((size_t)KB * MT + (size_t)b * MT + n) * 64; // sin plane
        for (int f = 0; f < 32; ++f) {
            const float p = x * Wr[f * 2] + y * Wr[f * 2 + 1];
            const float c = cosf(p), s = sinf(p);
            ec[2 * f] = c; ec[2 * f + 1] = c;
            es[2 * f] = s; es[2 * f + 1] = s;
        }
    }
}

// qkv [B*MT][768] (col = h*192 + d*3 + c) ->
//   RoPE(q),RoPE(k) f16 [B][H][MT][64];  v head-transposed f16 [B][H][64][MT]
__global__ void rope_repack_kernel(const float* __restrict__ qkv,
                                   const float* __restrict__ enc,
                                   _Float16* __restrict__ Q, _Float16* __restrict__ K,
                                   _Float16* __restrict__ Vt)
{
    const size_t idx = (size_t)blockIdx.x * blockDim.x + threadIdx.x; // B*MT*256
    const int d = idx & 63;
    const int h = (idx >> 6) & 3;
    const size_t bn = idx >> 8;
    const float* row = qkv + bn * 768;
    const int ci = h * 192 + d * 3;
    const int cp = h * 192 + (d ^ 1) * 3;
    const float q = row[ci], k = row[ci + 1], v = row[ci + 2];
    const float qp = row[cp], kp = row[cp + 1];
    const float sgn = (d & 1) ? 1.0f : -1.0f; // rotate_half
    const float c0 = enc[bn * 64 + d];
    const float s0 = enc[((size_t)KB * MT + bn) * 64 + d];
    const size_t b = bn >> 10, n = bn & 1023;
    const size_t o = (((size_t)b * NH + h) * MT + n) * 64 + d;
    Q[o] = (_Float16)(q * c0 + sgn * qp * s0);
    K[o] = (_Float16)(k * c0 + sgn * kp * s0);
    Vt[(((size_t)b * NH + h) * HDIM + d) * MT + n] = (_Float16)v;
}

// [B*MT][256] f32 -> [B][H][MT][64] f16
__global__ void repack_heads_kernel(const float* __restrict__ X, _Float16* __restrict__ Y)
{
    const size_t idx = (size_t)blockIdx.x * blockDim.x + threadIdx.x;
    const int c = idx & 255;
    const size_t bn = idx >> 8;
    const int h = c >> 6, d = c & 63;
    const size_t b = bn >> 10, n = bn & 1023;
    Y[(((size_t)b * NH + h) * MT + n) * 64 + d] = (_Float16)X[idx];
}

// [B*MT][256] f32 -> [B][H][64][MT] f16 (head-transposed, for V operands)
__global__ void repack_heads_t_kernel(const float* __restrict__ X, _Float16* __restrict__ Y)
{
    const size_t idx = (size_t)blockIdx.x * blockDim.x + threadIdx.x;
    const int c = idx & 255;
    const size_t bn = idx >> 8;
    const int h = c >> 6, d = c & 63;
    const size_t b = bn >> 10, n = bn & 1023;
    Y[(((size_t)b * NH + h) * HDIM + d) * MT + n] = (_Float16)X[idx];
}

// [B][H][MT][64] f32 -> [B*MT][256] f16
__global__ void unpack_heads_kernel(const float* __restrict__ Y, _Float16* __restrict__ X)
{
    const size_t idx = (size_t)blockIdx.x * blockDim.x + threadIdx.x;
    const int c = idx & 255;
    const size_t bn = idx >> 8;
    const int h = c >> 6, d = c & 63;
    const size_t b = bn >> 10, n = bn & 1023;
    X[idx] = (_Float16)Y[(((size_t)b * NH + h) * MT + n) * 64 + d];
}

// row softmax over f32 input; normalized probabilities written to f16 mirror
__global__ void softmax_rows_kernel(float* __restrict__ X, _Float16* __restrict__ Xh,
                                    int Cn)
{
    const int t = threadIdx.x;
    float* row = X + (size_t)blockIdx.x * Cn;
    _Float16* rowh = Xh + (size_t)blockIdx.x * Cn;
    __shared__ float red[256];
    float m = -3.4e38f;
    for (int j = t; j < Cn; j += 256) m = fmaxf(m, row[j]);
    red[t] = m; __syncthreads();
    for (int k = 128; k > 0; k >>= 1) { if (t < k) red[t] = fmaxf(red[t], red[t + k]); __syncthreads(); }
    m = red[0]; __syncthreads();
    float s = 0.0f;
    for (int j = t; j < Cn; j += 256) { float e = expf(row[j] - m); row[j] = e; s += e; }
    red[t] = s; __syncthreads();
    for (int k = 128; k > 0; k >>= 1) { if (t < k) red[t] += red[t + k]; __syncthreads(); }
    const float inv = 1.0f / red[0];
    for (int j = t; j < Cn; j += 256) rowh[j] = (_Float16)(row[j] * inv);
}

__global__ void lse_rows_kernel(const float* __restrict__ X, float* __restrict__ out, int Cn)
{
    const int t = threadIdx.x;
    const float* row = X + (size_t)blockIdx.x * Cn;
    __shared__ float red[256];
    float m = -3.4e38f;
    for (int j = t; j < Cn; j += 256) m = fmaxf(m, row[j]);
    red[t] = m; __syncthreads();
    for (int k = 128; k > 0; k >>= 1) { if (t < k) red[t] = fmaxf(red[t], red[t + k]); __syncthreads(); }
    m = red[0]; __syncthreads();
    float s = 0.0f;
    for (int j = t; j < Cn; j += 256) s += expf(row[j] - m);
    red[t] = s; __syncthreads();
    for (int k = 128; k > 0; k >>= 1) { if (t < k) red[t] += red[t + k]; __syncthreads(); }
    if (t == 0) out[blockIdx.x] = m + logf(red[0]);
}

// column logsumexp over sim [B][MT][MT]: one block per (b, j)
__global__ void lse_cols_kernel(const float* __restrict__ X, float* __restrict__ out)
{
    const int t = threadIdx.x;
    const int b = blockIdx.x >> 10, j = blockIdx.x & 1023;
    const float* base = X + (size_t)b * MT * MT + j;
    __shared__ float red[256];
    float m = -3.4e38f;
    for (int i = t; i < MT; i += 256) m = fmaxf(m, base[(size_t)i * MT]);
    red[t] = m; __syncthreads();
    for (int k = 128; k > 0; k >>= 1) { if (t < k) red[t] = fmaxf(red[t], red[t + k]); __syncthreads(); }
    m = red[0]; __syncthreads();
    float s = 0.0f;
    for (int i = t; i < MT; i += 256) s += expf(base[(size_t)i * MT] - m);
    red[t] = s; __syncthreads();
    for (int k = 128; k > 0; k >>= 1) { if (t < k) red[t] += red[t + k]; __syncthreads(); }
    if (t == 0) out[blockIdx.x] = m + logf(red[0]);
}

// LayerNorm (per row of Cn) + exact GELU; f32 in, f16 out
__global__ void ln_gelu_kernel(const float* __restrict__ X, const float* __restrict__ w,
                               const float* __restrict__ bb, _Float16* __restrict__ out,
                               int Cn)
{
    const int t = threadIdx.x;
    const float* row = X + (size_t)blockIdx.x * Cn;
    _Float16* orow = out + (size_t)blockIdx.x * Cn;
    __shared__ float red[256];
    float s = 0.0f;
    for (int j = t; j < Cn; j += 256) s += row[j];
    red[t] = s; __syncthreads();
    for (int k = 128; k > 0; k >>= 1) { if (t < k) red[t] += red[t + k]; __syncthreads(); }
    const float mu = red[0] / Cn; __syncthreads();
    float v = 0.0f;
    for (int j = t; j < Cn; j += 256) { float d = row[j] - mu; v += d * d; }
    red[t] = v; __syncthreads();
    for (int k = 128; k > 0; k >>= 1) { if (t < k) red[t] += red[t + k]; __syncthreads(); }
    const float inv = rsqrtf(red[0] / Cn + 1e-5f);
    for (int j = t; j < Cn; j += 256) {
        const float g = (row[j] - mu) * inv * w[j] + bb[j];
        orow[j] = (_Float16)(0.5f * g * (1.0f + erff(g * 0.70710678118f)));
    }
}

// cat[row][0:256]=x, cat[row][256:512]=msg  (f32 inputs -> f16 output)
__global__ void concat_kernel(const float* __restrict__ x, const float* __restrict__ msg,
                              _Float16* __restrict__ cat)
{
    const size_t idx = (size_t)blockIdx.x * blockDim.x + threadIdx.x;
    const int c = idx & 511;
    const size_t row = idx >> 9;
    cat[idx] = (_Float16)((c < DD) ? x[row * DD + c] : msg[row * DD + (c - DD)]);
}

// z[row] = dot(X[row], w) + b   (256 cols, one block per row)
__global__ void zdot_kernel(const float* __restrict__ X, const float* __restrict__ w,
                            const float* __restrict__ b, float* __restrict__ z)
{
    const int t = threadIdx.x;
    __shared__ float red[256];
    red[t] = X[(size_t)blockIdx.x * DD + t] * w[t];
    __syncthreads();
    for (int k = 128; k > 0; k >>= 1) { if (t < k) red[t] += red[t + k]; __syncthreads(); }
    if (t == 0) z[blockIdx.x] = red[0] + b[0];
}

// scores[b,i,j] = sim - lse_r[b,i] - lse_c[b,j] + lsig(z0[b,i]) + lsig(z1[b,j])
__global__ void assemble_scores_kernel(const float* __restrict__ sim,
                                       const float* __restrict__ lser,
                                       const float* __restrict__ lsec,
                                       const float* __restrict__ z0,
                                       const float* __restrict__ z1,
                                       float* __restrict__ scores)
{
    const size_t idx = (size_t)blockIdx.x * blockDim.x + threadIdx.x; // B*MT*MT
    const size_t b = idx >> 20;
    const size_t rem = idx & ((1u << 20) - 1);
    const size_t i = rem >> 10, j = rem & 1023;
    const float s = sim[idx] - lser[b * MT + i] - lsec[b * MT + j]
                  + lsig(z0[b * MT + i]) + lsig(z1[b * MT + j]);
    scores[b * 1025 * 1025 + i * 1025 + j] = s;
}

__global__ void borders_kernel(const float* __restrict__ z0, const float* __restrict__ z1,
                               float* __restrict__ scores)
{
    const size_t idx = (size_t)blockIdx.x * blockDim.x + threadIdx.x; // B*MT
    const size_t b = idx >> 10, i = idx & 1023;
    float* sb = scores + b * 1025 * 1025;
    sb[i * 1025 + 1024] = lsig(-z0[idx]);
    sb[(size_t)1024 * 1025 + i] = lsig(-z1[idx]);
    if (i == 0) sb[(size_t)1024 * 1025 + 1024] = 0.0f;
}

__global__ void rowargmax_kernel(const float* __restrict__ scores,
                                 int* __restrict__ am, float* __restrict__ mx)
{
    const int t = threadIdx.x;
    const int b = blockIdx.x >> 10, i = blockIdx.x & 1023;
    const float* row = scores + (size_t)b * 1025 * 1025 + (size_t)i * 1025;
    __shared__ float sv[256]; __shared__ int si[256];
    float best = -3.4e38f; int bi = 0;
    for (int j = t; j < MT; j += 256) { float v = row[j]; if (v > best) { best = v; bi = j; } }
    sv[t] = best; si[t] = bi; __syncthreads();
    for (int k = 128; k > 0; k >>= 1) {
        if (t < k && sv[t + k] > sv[t]) { sv[t] = sv[t + k]; si[t] = si[t + k]; }
        __syncthreads();
    }
    if (t == 0) { am[blockIdx.x] = si[0]; mx[blockIdx.x] = sv[0]; }
}

__global__ void colargmax_kernel(const float* __restrict__ scores, int* __restrict__ am)
{
    const int t = threadIdx.x;
    const int b = blockIdx.x >> 10, j = blockIdx.x & 1023;
    const float* base = scores + (size_t)b * 1025 * 1025 + j;
    __shared__ float sv[256]; __shared__ int si[256];
    float best = -3.4e38f; int bi = 0;
    for (int i = t; i < MT; i += 256) { float v = base[(size_t)i * 1025]; if (v > best) { best = v; bi = i; } }
    sv[t] = best; si[t] = bi; __syncthreads();
    for (int k = 128; k > 0; k >>= 1) {
        if (t < k && sv[t + k] > sv[t]) { sv[t] = sv[t + k]; si[t] = si[t + k]; }
        __syncthreads();
    }
    if (t == 0) am[blockIdx.x] = si[0];
}

__global__ void filter_i_kernel(const int* __restrict__ m0i, const int* __restrict__ m1i,
                                const float* __restrict__ mx0, float* __restrict__ msc0,
                                int* __restrict__ vflag, float* __restrict__ om0,
                                float* __restrict__ oms0)
{
    const size_t idx = (size_t)blockIdx.x * blockDim.x + threadIdx.x; // B*MT
    const size_t b = idx >> 10, i = idx & 1023;
    const int mv = m0i[idx];
    const bool mut = (m1i[b * MT + mv] == (int)i);
    const float ms = mut ? expf(mx0[idx]) : 0.0f;
    const int valid = (mut && ms > 0.1f) ? 1 : 0;
    msc0[idx] = ms; vflag[idx] = valid;
    om0[idx] = valid ? (float)mv : -1.0f;
    oms0[idx] = ms;
}

__global__ void filter_j_kernel(const int* __restrict__ m0i, const int* __restrict__ m1i,
                                const float* __restrict__ msc0, const int* __restrict__ vflag,
                                float* __restrict__ om1, float* __restrict__ oms1)
{
    const size_t idx = (size_t)blockIdx.x * blockDim.x + threadIdx.x; // B*MT
    const size_t b = idx >> 10, j = idx & 1023;
    const int mv = m1i[idx];
    const bool mut = (m0i[b * MT + mv] == (int)j);
    const float ms = mut ? msc0[b * MT + mv] : 0.0f;
    const int valid = (mut && vflag[b * MT + mv]) ? 1 : 0;
    om1[idx] = valid ? (float)mv : -1.0f;
    oms1[idx] = ms;
}

// ---------------------------------------------------------------------------
extern "C" void kernel_launch(void* const* d_in, const int* in_sizes, int n_in,
                              void* d_out, int out_size, void* d_ws, size_t ws_size,
                              hipStream_t stream)
{
    (void)in_sizes; (void)n_in; (void)out_size; (void)ws_size;
    typedef long long ll;

    const float* kpts0 = (const float*)d_in[0];
    const float* kpts1 = (const float*)d_in[1];
    const float* desc0 = (const float*)d_in[2];
    const float* desc1 = (const float*)d_in[3];
    const float* Wr    = (const float*)d_in[4];
    const float* sa_qkv_w = (const float*)d_in[5];
    const float* sa_qkv_b = (const float*)d_in[6];
    const float* sa_o_b   = (const float*)d_in[8];
    const float* sa_f1_b  = (const float*)d_in[10];
    const float* sa_ln_w  = (const float*)d_in[11];
    const float* sa_ln_b  = (const float*)d_in[12];
    const float* sa_f2_b  = (const float*)d_in[14];
    const float* ca_qk_b  = (const float*)d_in[16];
    const float* ca_v_b   = (const float*)d_in[18];
    const float* ca_o_b   = (const float*)d_in[20];
    const float* ca_f1_b  = (const float*)d_in[22];
    const float* ca_ln_w  = (const float*)d_in[23];
    const float* ca_ln_b  = (const float*)d_in[24];
    const float* ca_f2_b  = (const float*)d_in[26];
    const float* proj_b   = (const float*)d_in[28];
    const float* match_w  = (const float*)d_in[29];
    const float* match_b  = (const float*)d_in[30];

    // ---------------- workspace carve (16B aligned) ----------------
    char* base = (char*)d_ws;
    size_t off = 0;
    auto takeF = [&](size_t n) { off = (off + 15) & ~(size_t)15;
                                 float* p = (float*)(base + off); off += n * 4; return p; };
    auto takeH = [&](size_t n) { off = (off + 15) & ~(size_t)15;
                                 _Float16* p = (_Float16*)(base + off); off += n * 2; return p; };

    // f32 buffers
    float* enc0  = takeF((size_t)2 * KB * MT * HDIM);
    float* enc1  = takeF((size_t)2 * KB * MT * HDIM);
    float* d0v   = takeF((size_t)KB * MT * DD);
    float* d1v   = takeF((size_t)KB * MT * DD);
    float* qkvb  = takeF((size_t)KB * MT * 3 * DD);
    float* attnA = takeF((size_t)KB * NH * MT * MT);
    float* attnB = takeF((size_t)KB * NH * MT * MT);
    float* ctxA  = takeF((size_t)KB * NH * MT * HDIM);
    float* ctxB  = takeF((size_t)KB * NH * MT * HDIM);
    float* flatA = takeF((size_t)KB * MT * DD);
    float* msgA  = takeF((size_t)KB * MT * DD);
    float* msgB  = takeF((size_t)KB * MT * DD);
    float* hbuf  = takeF((size_t)KB * MT * 2 * DD);
    float* simAs = takeF((size_t)KB * MT * MT);
    float* z0b   = takeF((size_t)KB * MT);
    float* z1b   = takeF((size_t)KB * MT);
    float* lserb = takeF((size_t)KB * MT);
    float* lsecb = takeF((size_t)KB * MT);
    float* mx0b  = takeF((size_t)KB * MT);
    float* msc0b = takeF((size_t)KB * MT);
    int*   m0ib  = (int*)takeF((size_t)KB * MT);
    int*   m1ib  = (int*)takeF((size_t)KB * MT);
    int*   vflgb = (int*)takeF((size_t)KB * MT);

    // f16 activation buffers
    _Float16* d0h   = takeH((size_t)KB * MT * DD);
    _Float16* d1h   = takeH((size_t)KB * MT * DD);
    _Float16* Qh    = takeH((size_t)KB * NH * MT * HDIM);
    _Float16* Kh    = takeH((size_t)KB * NH * MT * HDIM);
    _Float16* Vt    = takeH((size_t)KB * NH * HDIM * MT);
    _Float16* V2t   = takeH((size_t)KB * NH * HDIM * MT);
    _Float16* attnAh = takeH((size_t)KB * NH * MT * MT);
    _Float16* attnBh = takeH((size_t)KB * NH * MT * MT);
    _Float16* uflAh = takeH((size_t)KB * MT * DD);
    _Float16* uflBh = takeH((size_t)KB * MT * DD);
    _Float16* cath  = takeH((size_t)KB * MT * 2 * DD);
    _Float16* hbufh = takeH((size_t)KB * MT * 2 * DD);
    _Float16* md0h  = takeH((size_t)KB * MT * DD);
    _Float16* md1h  = takeH((size_t)KB * MT * DD);

    // f16 weight copies
    _Float16* wqkvh = takeH((size_t)NL * 3 * DD * DD);
    _Float16* wsoh  = takeH((size_t)NL * DD * DD);
    _Float16* wsf1h = takeH((size_t)NL * 2 * DD * 2 * DD);
    _Float16* wsf2h = takeH((size_t)NL * DD * 2 * DD);
    _Float16* wcqkh = takeH((size_t)NL * DD * DD);
    _Float16* wcvh  = takeH((size_t)NL * DD * DD);
    _Float16* wcoh  = takeH((size_t)NL * DD * DD);
    _Float16* wcf1h = takeH((size_t)NL * 2 * DD * 2 * DD);
    _Float16* wcf2h = takeH((size_t)NL * DD * 2 * DD);
    _Float16* wprjh = takeH((size_t)DD * DD);

    // ---------------- output carve ----------------
    float* out     = (float*)d_out;
    float* out_sc  = out;                                    // B*1025*1025
    float* out_m0  = out + (size_t)KB * 1025 * 1025;
    float* out_m1  = out_m0 + (size_t)KB * MT;
    float* out_ms0 = out_m1 + (size_t)KB * MT;
    float* out_ms1 = out_ms0 + (size_t)KB * MT;

    auto cvt = [&](const float* X, _Float16* Y, size_t n) {
        cvt_f16_kernel<<<(unsigned)((n + 255) / 256), 256, 0, stream>>>(X, Y, n);
    };
    auto gemm = [&](const _Float16* A, const _Float16* Bm, const float* bias,
                    const float* resid, float* C, _Float16* Ch,
                    int Ms, int Ns, int Ks, ll sA, ll sB, ll sC, int nb, float scale) {
        dim3 g((unsigned)(Ms / 16), (unsigned)(Ns / 64), (unsigned)nb);
        wmma_gemm_kernel<<<g, dim3(32), 0, stream>>>(A, Bm, bias, resid, C, Ch,
                                                     Ms, Ns, Ks, sA, sB, sC, scale);
    };

    const int ROWS  = KB * MT;                 // 2048
    const int EL256 = (KB * MT * DD) / 256;    // 2048 blocks of 256 threads

    // ---- one-time (per call) conversions: weights + initial descriptors ----
    cvt((const float*)d_in[5],  wqkvh, (size_t)NL * 3 * DD * DD);
    cvt((const float*)d_in[7],  wsoh,  (size_t)NL * DD * DD);
    cvt((const float*)d_in[9],  wsf1h, (size_t)NL * 2 * DD * 2 * DD);
    cvt((const float*)d_in[13], wsf2h, (size_t)NL * DD * 2 * DD);
    cvt((const float*)d_in[15], wcqkh, (size_t)NL * DD * DD);
    cvt((const float*)d_in[17], wcvh,  (size_t)NL * DD * DD);
    cvt((const float*)d_in[19], wcoh,  (size_t)NL * DD * DD);
    cvt((const float*)d_in[21], wcf1h, (size_t)NL * 2 * DD * 2 * DD);
    cvt((const float*)d_in[25], wcf2h, (size_t)NL * DD * 2 * DD);
    cvt((const float*)d_in[27], wprjh, (size_t)DD * DD);
    cvt(desc0, d0h, (size_t)ROWS * DD);
    cvt(desc1, d1h, (size_t)ROWS * DD);
    hipMemcpyAsync(d0v, desc0, sizeof(float) * (size_t)ROWS * DD, hipMemcpyDeviceToDevice, stream);
    hipMemcpyAsync(d1v, desc1, sizeof(float) * (size_t)ROWS * DD, hipMemcpyDeviceToDevice, stream);

    posenc_kernel<<<KB, 256, 0, stream>>>(kpts0, Wr, enc0);
    posenc_kernel<<<KB, 256, 0, stream>>>(kpts1, Wr, enc1);

    const float iscl = 0.125f; // 1/sqrt(HD)

    auto ffn = [&](float* x, _Float16* xh, const float* msg,
                   const _Float16* f1w, const float* f1b,
                   const float* lnw, const float* lnb,
                   const _Float16* f2w, const float* f2b) {
        concat_kernel<<<(ROWS * 2 * DD) / 256, 256, 0, stream>>>(x, msg, cath);
        gemm(cath, f1w, f1b, nullptr, hbuf, nullptr, ROWS, 2 * DD, 2 * DD, 0, 0, 0, 1, 1.0f);
        ln_gelu_kernel<<<ROWS, 256, 0, stream>>>(hbuf, lnw, lnb, hbufh, 2 * DD);
        gemm(hbufh, f2w, f2b, x, x, xh, ROWS, DD, 2 * DD, 0, 0, 0, 1, 1.0f);
    };

    for (int l = 0; l < NL; ++l) {
        const _Float16* qkvw = wqkvh + (size_t)l * 3 * DD * DD;
        const float* qkvbs = sa_qkv_b + (size_t)l * 3 * DD;
        const _Float16* sow = wsoh + (size_t)l * DD * DD;
        const float* sob = sa_o_b + (size_t)l * DD;
        const _Float16* sf1w = wsf1h + (size_t)l * 2 * DD * 2 * DD;
        const float* sf1b = sa_f1_b + (size_t)l * 2 * DD;
        const float* slnw = sa_ln_w + (size_t)l * 2 * DD;
        const float* slnb = sa_ln_b + (size_t)l * 2 * DD;
        const _Float16* sf2w = wsf2h + (size_t)l * DD * 2 * DD;
        const float* sf2b = sa_f2_b + (size_t)l * DD;

        // ---- self attention on each image ----
        for (int img = 0; img < 2; ++img) {
            float* x = img ? d1v : d0v;
            _Float16* xh = img ? d1h : d0h;
            const float* enc = img ? enc1 : enc0;
            gemm(xh, qkvw, qkvbs, nullptr, qkvb, nullptr, ROWS, 3 * DD, DD, 0, 0, 0, 1, 1.0f);
            rope_repack_kernel<<<EL256, 256, 0, stream>>>(qkvb, enc, Qh, Kh, Vt);
            gemm(Qh, Kh, nullptr, nullptr, attnA, nullptr, MT, MT, HDIM,
                 (ll)MT * HDIM, (ll)MT * HDIM, (ll)MT * MT, KB * NH, iscl);
            softmax_rows_kernel<<<KB * NH * MT, 256, 0, stream>>>(attnA, attnAh, MT);
            gemm(attnAh, Vt, nullptr, nullptr, ctxA, nullptr, MT, HDIM, MT,
                 (ll)MT * MT, (ll)HDIM * MT, (ll)MT * HDIM, KB * NH, 1.0f);
            unpack_heads_kernel<<<EL256, 256, 0, stream>>>(ctxA, uflAh);
            gemm(uflAh, sow, sob, nullptr, msgA, nullptr, ROWS, DD, DD, 0, 0, 0, 1, 1.0f);
            ffn(x, xh, msgA, sf1w, sf1b, slnw, slnb, sf2w, sf2b);
        }

        // ---- cross attention ----
        const _Float16* qkw = wcqkh + (size_t)l * DD * DD;
        const float* qkb = ca_qk_b + (size_t)l * DD;
        const _Float16* vw = wcvh + (size_t)l * DD * DD;
        const float* vb = ca_v_b + (size_t)l * DD;
        const _Float16* cow = wcoh + (size_t)l * DD * DD;
        const float* cob = ca_o_b + (size_t)l * DD;
        const _Float16* cf1w = wcf1h + (size_t)l * 2 * DD * 2 * DD;
        const float* cf1b = ca_f1_b + (size_t)l * 2 * DD;
        const float* clnw = ca_ln_w + (size_t)l * 2 * DD;
        const float* clnb = ca_ln_b + (size_t)l * 2 * DD;
        const _Float16* cf2w = wcf2h + (size_t)l * DD * 2 * DD;
        const float* cf2b = ca_f2_b + (size_t)l * DD;

        gemm(d0h, qkw, qkb, nullptr, flatA, nullptr, ROWS, DD, DD, 0, 0, 0, 1, 1.0f);
        repack_heads_kernel<<<EL256, 256, 0, stream>>>(flatA, Qh);    // qk0
        gemm(d1h, qkw, qkb, nullptr, flatA, nullptr, ROWS, DD, DD, 0, 0, 0, 1, 1.0f);
        repack_heads_kernel<<<EL256, 256, 0, stream>>>(flatA, Kh);    // qk1
        gemm(d0h, vw, vb, nullptr, flatA, nullptr, ROWS, DD, DD, 0, 0, 0, 1, 1.0f);
        repack_heads_t_kernel<<<EL256, 256, 0, stream>>>(flatA, Vt);  // v0 (transposed)
        gemm(d1h, vw, vb, nullptr, flatA, nullptr, ROWS, DD, DD, 0, 0, 0, 1, 1.0f);
        repack_heads_t_kernel<<<EL256, 256, 0, stream>>>(flatA, V2t); // v1 (transposed)

        gemm(Qh, Kh, nullptr, nullptr, attnA, nullptr, MT, MT, HDIM,
             (ll)MT * HDIM, (ll)MT * HDIM, (ll)MT * MT, KB * NH, iscl);       // sim
        gemm(Kh, Qh, nullptr, nullptr, attnB, nullptr, MT, MT, HDIM,
             (ll)MT * HDIM, (ll)MT * HDIM, (ll)MT * MT, KB * NH, iscl);       // sim^T
        softmax_rows_kernel<<<KB * NH * MT, 256, 0, stream>>>(attnA, attnAh, MT);
        softmax_rows_kernel<<<KB * NH * MT, 256, 0, stream>>>(attnB, attnBh, MT);
        gemm(attnAh, V2t, nullptr, nullptr, ctxA, nullptr, MT, HDIM, MT,
             (ll)MT * MT, (ll)HDIM * MT, (ll)MT * HDIM, KB * NH, 1.0f);       // m0 ctx
        gemm(attnBh, Vt, nullptr, nullptr, ctxB, nullptr, MT, HDIM, MT,
             (ll)MT * MT, (ll)HDIM * MT, (ll)MT * HDIM, KB * NH, 1.0f);       // m1 ctx
        unpack_heads_kernel<<<EL256, 256, 0, stream>>>(ctxA, uflAh);
        unpack_heads_kernel<<<EL256, 256, 0, stream>>>(ctxB, uflBh);
        gemm(uflAh, cow, cob, nullptr, msgA, nullptr, ROWS, DD, DD, 0, 0, 0, 1, 1.0f);
        gemm(uflBh, cow, cob, nullptr, msgB, nullptr, ROWS, DD, DD, 0, 0, 0, 1, 1.0f);
        ffn(d0v, d0h, msgA, cf1w, cf1b, clnw, clnb, cf2w, cf2b);
        ffn(d1v, d1h, msgB, cf1w, cf1b, clnw, clnb, cf2w, cf2b);
    }

    // ---- assignment ----
    gemm(d0h, wprjh, proj_b, nullptr, nullptr, md0h, ROWS, DD, DD, 0, 0, 0, 1, 0.25f);
    gemm(d1h, wprjh, proj_b, nullptr, nullptr, md1h, ROWS, DD, DD, 0, 0, 0, 1, 0.25f);
    gemm(md0h, md1h, nullptr, nullptr, simAs, nullptr, MT, MT, DD,
         (ll)MT * DD, (ll)MT * DD, (ll)MT * MT, KB, 1.0f);
    zdot_kernel<<<ROWS, 256, 0, stream>>>(d0v, match_w, match_b, z0b);
    zdot_kernel<<<ROWS, 256, 0, stream>>>(d1v, match_w, match_b, z1b);
    lse_rows_kernel<<<ROWS, 256, 0, stream>>>(simAs, lserb, MT);
    lse_cols_kernel<<<ROWS, 256, 0, stream>>>(simAs, lsecb);
    assemble_scores_kernel<<<(KB * MT * MT) / 256, 256, 0, stream>>>(
        simAs, lserb, lsecb, z0b, z1b, out_sc);
    borders_kernel<<<ROWS / 256, 256, 0, stream>>>(z0b, z1b, out_sc);

    // ---- match filtering ----
    rowargmax_kernel<<<ROWS, 256, 0, stream>>>(out_sc, m0ib, mx0b);
    colargmax_kernel<<<ROWS, 256, 0, stream>>>(out_sc, m1ib);
    filter_i_kernel<<<ROWS / 256, 256, 0, stream>>>(m0ib, m1ib, mx0b, msc0b, vflgb,
                                                    out_m0, out_ms0);
    filter_j_kernel<<<ROWS / 256, 256, 0, stream>>>(m0ib, m1ib, msc0b, vflgb,
                                                    out_m1, out_ms1);
}
